// LSTMAutoEncoder_33844342293312
// MI455X (gfx1250) — matrix operand order repaired and checked by reading, not compiled, not run on hardware
//
#include <hip/hip_runtime.h>
#include <hip/hip_bf16.h>

typedef __attribute__((ext_vector_type(16))) __bf16 v16bf;
typedef __attribute__((ext_vector_type(8)))  float  v8f;

// ---------------------------------------------------------------------------
// Fragment loaders (wave32 WMMA; contiguous k order k0+16*(lane/16)+s, same in A and B)
// A: 16x32 bf16, row-major source. lane<16: row=lane, K = k0+[0..15]
//                                  lane>=16: row=lane-16, K = k0+16+[0..15]
// ---------------------------------------------------------------------------
__device__ __forceinline__ v16bf load_a_frag(const __bf16* __restrict__ A,
                                             int row0, int k0, int ld) {
  int lane = threadIdx.x & 31;
  int m    = lane & 15;
  int kb   = (lane & 16) ? 16 : 0;
  const __bf16* p = A + (size_t)(row0 + m) * ld + (k0 + kb);
  v16bf a;
#pragma unroll
  for (int j = 0; j < 8; ++j) { a[j] = p[j]; a[8 + j] = p[8 + j]; }
  return a;
}

// B: 32x16 bf16 taken from a PRE-TRANSPOSED weight WT[N][K] (row-major, ld=K).
// lane<16: K = k0+[0..15]; lane>=16: K = k0+16+[0..15]; column = n0 + (lane&15).
__device__ __forceinline__ v16bf load_b_fragT(const __bf16* __restrict__ WT,
                                              int k0, int n0, int ldk) {
  int lane = threadIdx.x & 31;
  int n    = lane & 15;
  int kb   = (lane & 16) ? 16 : 0;
  const __bf16* p = WT + (size_t)(n0 + n) * ldk + (k0 + kb);
  v16bf b;
#pragma unroll
  for (int j = 0; j < 16; ++j) b[j] = p[j];
  return b;
}

__device__ __forceinline__ float sigmoidf_(float x) {
  return 1.0f / (1.0f + __expf(-x));
}

// ---------------------------------------------------------------------------
// Fused LSTM cell: z = A1@W1T' (+ A2@W2T') + bias ; gates ; update c ; emit h.
// Each wave: one 16x16 position x 4 gates (cols col0+{0,512,1024,1536} of z).
// Grid: 64 blocks x 256 threads (8 waves) -> 512 wave tiles = (256/16)x(512/16).
// use_prev_c=1: c = sig(f)*c_old + sig(i)*relu(g)   (decoder cell)
// use_prev_c=0: c = sig(i)*relu(g)                  (encoder layer)
// h = sig(o)*relu(c)
// ---------------------------------------------------------------------------
__global__ __launch_bounds__(256) void lstm_cell_kernel(
    const __bf16* __restrict__ A1, int K1, const __bf16* __restrict__ W1T,
    const __bf16* __restrict__ A2, int K2, const __bf16* __restrict__ W2T,
    const float* __restrict__ bias,
    float* __restrict__ c_io,
    __bf16* __restrict__ h_out,
    int use_prev_c) {
  int wave = threadIdx.x >> 5;
  int tile = blockIdx.x * 8 + wave;     // 0..511
  int row0 = (tile >> 5) << 4;          // 16 row tiles (B=256)
  int col0 = (tile & 31) << 4;          // 32 col tiles (H=512)

  v8f acc0 = {}, acc1 = {}, acc2 = {}, acc3 = {};

  for (int k = 0; k < K1; k += 32) {
    v16bf a = load_a_frag(A1, row0, k, K1);
    v16bf b0 = load_b_fragT(W1T, k, 0    + col0, K1);
    v16bf b1 = load_b_fragT(W1T, k, 512  + col0, K1);
    v16bf b2 = load_b_fragT(W1T, k, 1024 + col0, K1);
    v16bf b3 = load_b_fragT(W1T, k, 1536 + col0, K1);
    acc0 = __builtin_amdgcn_wmma_f32_16x16x32_bf16(false, a, false, b0, (short)0, acc0, false, false);
    acc1 = __builtin_amdgcn_wmma_f32_16x16x32_bf16(false, a, false, b1, (short)0, acc1, false, false);
    acc2 = __builtin_amdgcn_wmma_f32_16x16x32_bf16(false, a, false, b2, (short)0, acc2, false, false);
    acc3 = __builtin_amdgcn_wmma_f32_16x16x32_bf16(false, a, false, b3, (short)0, acc3, false, false);
  }
  if (A2 != nullptr) {
    for (int k = 0; k < K2; k += 32) {
      v16bf a = load_a_frag(A2, row0, k, K2);
      v16bf b0 = load_b_fragT(W2T, k, 0    + col0, K2);
      v16bf b1 = load_b_fragT(W2T, k, 512  + col0, K2);
      v16bf b2 = load_b_fragT(W2T, k, 1024 + col0, K2);
      v16bf b3 = load_b_fragT(W2T, k, 1536 + col0, K2);
      acc0 = __builtin_amdgcn_wmma_f32_16x16x32_bf16(false, a, false, b0, (short)0, acc0, false, false);
      acc1 = __builtin_amdgcn_wmma_f32_16x16x32_bf16(false, a, false, b1, (short)0, acc1, false, false);
      acc2 = __builtin_amdgcn_wmma_f32_16x16x32_bf16(false, a, false, b2, (short)0, acc2, false, false);
      acc3 = __builtin_amdgcn_wmma_f32_16x16x32_bf16(false, a, false, b3, (short)0, acc3, false, false);
    }
  }

  // Epilogue: C/D layout -> lane holds col = col0+(lane&15), rows rbase+r
  int lane  = threadIdx.x & 31;
  int col   = col0 + (lane & 15);
  int rbase = row0 + ((lane & 16) ? 8 : 0);
  float bi = bias[col];
  float bf = bias[512 + col];
  float bg = bias[1024 + col];
  float bo = bias[1536 + col];
#pragma unroll
  for (int r = 0; r < 8; ++r) {
    int row = rbase + r;
    size_t idx = (size_t)row * 512 + col;
    float zi = acc0[r] + bi;
    float zf = acc1[r] + bf;
    float zg = acc2[r] + bg;
    float zo = acc3[r] + bo;
    float gi = sigmoidf_(zi);
    float go = sigmoidf_(zo);
    float gg = fmaxf(zg, 0.0f);
    float c  = gi * gg;
    if (use_prev_c) c += sigmoidf_(zf) * c_io[idx];
    c_io[idx]  = c;                       // each element owned by one lane reg
    h_out[idx] = (__bf16)(go * fmaxf(c, 0.0f));
  }
}

// ---------------------------------------------------------------------------
// y = H(256x512,bf16) @ WdT'(512x64) + bdense ; scatter into output slot t and
// refresh bf16 x-buffer. Grid: 8 blocks x 256 thr -> 64 wave tiles (16x4).
// ---------------------------------------------------------------------------
__global__ __launch_bounds__(256) void dense_kernel(
    const __bf16* __restrict__ H, const __bf16* __restrict__ WdT,
    const float* __restrict__ bias,
    float* __restrict__ Yout, int yRowStride,
    __bf16* __restrict__ Xout) {
  int wave = threadIdx.x >> 5;
  int tile = blockIdx.x * 8 + wave;     // 0..63
  int row0 = (tile >> 2) << 4;          // 16 row tiles
  int col0 = (tile & 3) << 4;           // 4 col tiles (D=64)

  v8f acc = {};
  for (int k = 0; k < 512; k += 32) {
    v16bf a = load_a_frag(H, row0, k, 512);
    v16bf b = load_b_fragT(WdT, k, col0, 512);
    acc = __builtin_amdgcn_wmma_f32_16x16x32_bf16(false, a, false, b, (short)0, acc, false, false);
  }
  int lane  = threadIdx.x & 31;
  int col   = col0 + (lane & 15);
  int rbase = row0 + ((lane & 16) ? 8 : 0);
  float bb = bias[col];
#pragma unroll
  for (int r = 0; r < 8; ++r) {
    int row = rbase + r;
    float y = acc[r] + bb;
    Yout[(size_t)row * yRowStride + col] = y;
    Xout[(size_t)row * 64 + col] = (__bf16)y;
  }
}

// ---------------------------------------------------------------------------
// One-time converters: fp32 [K][N] -> bf16 transposed [N][K]; last-timestep slice.
// ---------------------------------------------------------------------------
__global__ void transpose_to_bf16(const float* __restrict__ in,
                                  __bf16* __restrict__ out, int K, int N) {
  int idx = blockIdx.x * blockDim.x + threadIdx.x;
  if (idx >= K * N) return;
  int n = idx / K;
  int k = idx - n * K;
  out[idx] = (__bf16)in[(size_t)k * N + n];
}

__global__ void extract_last_ts(const float* __restrict__ enc,
                                __bf16* __restrict__ out) {
  int idx = blockIdx.x * blockDim.x + threadIdx.x;   // 256*64
  if (idx >= 256 * 64) return;
  int b = idx >> 6;
  int d = idx & 63;
  out[idx] = (__bf16)enc[(size_t)b * (128 * 64) + 127 * 64 + d];
}

// ---------------------------------------------------------------------------
extern "C" void kernel_launch(void* const* d_in, const int* in_sizes, int n_in,
                              void* d_out, int out_size, void* d_ws, size_t ws_size,
                              hipStream_t stream) {
  (void)in_sizes; (void)n_in; (void)out_size; (void)ws_size;
  const float* enc_in = (const float*)d_in[0];
  // d_in[1] decoder_inputs: unused by reference. d_in[3]/d_in[6] Ue0/Ue1: unused.
  const float* We0    = (const float*)d_in[2];
  const float* be0    = (const float*)d_in[4];
  const float* We1    = (const float*)d_in[5];
  const float* be1    = (const float*)d_in[7];
  const float* Wd0    = (const float*)d_in[8];
  const float* Ud0    = (const float*)d_in[9];
  const float* bd0    = (const float*)d_in[10];
  const float* Wd1    = (const float*)d_in[11];
  const float* Ud1    = (const float*)d_in[12];
  const float* bd1    = (const float*)d_in[13];
  const float* Wdense = (const float*)d_in[14];
  const float* bdense = (const float*)d_in[15];
  float* out = (float*)d_out;

  char* ws = (char*)d_ws;
  size_t off = 0;
  auto carve = [&](size_t bytes) -> char* {
    char* p = ws + off;
    off = (off + bytes + 255) & ~(size_t)255;
    return p;
  };
  __bf16* WTe0 = (__bf16*)carve((size_t)2048 * 64 * 2);
  __bf16* WTe1 = (__bf16*)carve((size_t)2048 * 512 * 2);
  __bf16* WTd0 = (__bf16*)carve((size_t)2048 * 64 * 2);
  __bf16* UTd0 = (__bf16*)carve((size_t)2048 * 512 * 2);
  __bf16* WTd1 = (__bf16*)carve((size_t)2048 * 64 * 2);
  __bf16* UTd1 = (__bf16*)carve((size_t)2048 * 512 * 2);
  __bf16* WdT  = (__bf16*)carve((size_t)64 * 512 * 2);
  __bf16* xbf  = (__bf16*)carve((size_t)256 * 64 * 2);
  __bf16* h0   = (__bf16*)carve((size_t)256 * 512 * 2);
  __bf16* h1   = (__bf16*)carve((size_t)256 * 512 * 2);
  __bf16* henc = (__bf16*)carve((size_t)256 * 512 * 2);
  float*  cbuf = (float*)carve((size_t)256 * 512 * 4);

  const int TPB = 256;
  transpose_to_bf16<<<(64 * 2048 + 255) / 256, TPB, 0, stream>>>(We0, WTe0, 64, 2048);
  transpose_to_bf16<<<(512 * 2048 + 255) / 256, TPB, 0, stream>>>(We1, WTe1, 512, 2048);
  transpose_to_bf16<<<(64 * 2048 + 255) / 256, TPB, 0, stream>>>(Wd0, WTd0, 64, 2048);
  transpose_to_bf16<<<(512 * 2048 + 255) / 256, TPB, 0, stream>>>(Ud0, UTd0, 512, 2048);
  transpose_to_bf16<<<(64 * 2048 + 255) / 256, TPB, 0, stream>>>(Wd1, WTd1, 64, 2048);
  transpose_to_bf16<<<(512 * 2048 + 255) / 256, TPB, 0, stream>>>(Ud1, UTd1, 512, 2048);
  transpose_to_bf16<<<(512 * 64 + 255) / 256, TPB, 0, stream>>>(Wdense, WdT, 512, 64);
  extract_last_ts<<<64, TPB, 0, stream>>>(enc_in, xbf);

  // Encoder collapses to last timestep only.
  lstm_cell_kernel<<<64, TPB, 0, stream>>>(xbf, 64, WTe0, nullptr, 0, nullptr,
                                           be0, cbuf, henc, 0);
  lstm_cell_kernel<<<64, TPB, 0, stream>>>(henc, 512, WTe1, nullptr, 0, nullptr,
                                           be1, cbuf, h0, 0);
  // x0 = enc_h @ Wdense + bdense -> output slot t=254 (sequence is reversed).
  dense_kernel<<<8, TPB, 0, stream>>>(h0, WdT, bdense, out + 254 * 64, 255 * 64, xbf);

  const __bf16* WTd[2] = {WTd0, WTd1};
  const __bf16* UTd[2] = {UTd0, UTd1};
  const float*  bd[2]  = {bd0, bd1};
  __bf16* hbuf[2] = {h0, h1};
  int cur = 0;  // h currently lives in hbuf[cur]
  for (int s = 0; s < 127; ++s) {
    for (int j = 0; j < 2; ++j) {
      __bf16* hin  = hbuf[cur];
      __bf16* hout = hbuf[cur ^ 1];
      lstm_cell_kernel<<<64, TPB, 0, stream>>>(xbf, 64, WTd[j], hin, 512, UTd[j],
                                               bd[j], cbuf, hout, 1);
      int t_out = 253 - 2 * s - j;  // reversed sequence position
      dense_kernel<<<8, TPB, 0, stream>>>(hout, WdT, bdense, out + t_out * 64,
                                          255 * 64, xbf);
      cur ^= 1;
    }
  }
}
